// CNNLSTMWithDropoutCell_19456201851160
// MI455X (gfx1250) — matrix operand order repaired and checked by reading, not compiled, not run on hardware
//
#include <hip/hip_runtime.h>

typedef __attribute__((ext_vector_type(16))) _Float16 v16h;
typedef __attribute__((ext_vector_type(8)))  _Float16 v8h;
typedef __attribute__((ext_vector_type(8)))  float    v8f;

#define B_ROWS 262144
#define DH     128     // D + H
#define NG     256     // 4*H gate columns

// ---------------------------------------------------------------------------
// Prep: transpose/concat the 4 fp32 weight matrices [128,64] into f16
// Wt[n][k], n = gate*64 + col, k = 0..127, row-major [256][128] in d_ws.
// ---------------------------------------------------------------------------
__global__ void prep_weights(const float* __restrict__ Wi, const float* __restrict__ Wf,
                             const float* __restrict__ Wo, const float* __restrict__ Wg,
                             _Float16* __restrict__ Wt) {
  int idx = blockIdx.x * blockDim.x + threadIdx.x;     // 0 .. 32767
  if (idx >= NG * DH) return;
  int n = idx >> 7;
  int k = idx & 127;
  const float* W = (n < 64) ? Wi : (n < 128) ? Wf : (n < 192) ? Wo : Wg;
  Wt[n * DH + k] = (_Float16)W[k * 64 + (n & 63)];
}

// Fast activations: v_exp_f32 + v_rcp_f32 (trans ops, co-execute with VALU).
// Avoids the ~9-instruction IEEE f32 divide expansion.
__device__ __forceinline__ float sigmoid_f(float x) {
  return __builtin_amdgcn_rcpf(1.0f + __expf(-x));
}
__device__ __forceinline__ float tanh_f(float x) {
  return __builtin_fmaf(2.0f, __builtin_amdgcn_rcpf(1.0f + __expf(-2.0f * x)), -1.0f);
}

// ---------------------------------------------------------------------------
// Main fused LSTM cell. 256 threads = 8 waves; each wave: 16 batch rows,
// 16 accumulator tiles (16x16 f32), K=128 via 4 x WMMA f32_16x16x32_f16.
// ---------------------------------------------------------------------------
__global__ __launch_bounds__(256) void lstm_cell(
    const float* __restrict__ x_t, const float* __restrict__ h_prev,
    const float* __restrict__ c_prev, const float* __restrict__ mask,
    const float* __restrict__ b_i, const float* __restrict__ b_f,
    const float* __restrict__ b_o, const float* __restrict__ b_g,
    const _Float16* __restrict__ Wt, float* __restrict__ out) {
  extern __shared__ _Float16 ldsW[];   // [256][128] f16 = 64 KB

  const int tid  = threadIdx.x;
  const int lane = tid & 31;
  const int wave = tid >> 5;
  const int hi   = (lane >> 4) & 1;    // lane group (0: lanes 0-15, 1: 16-31)
  const int colSel = lane & 15;
  const int rowBase = blockIdx.x * 128 + wave * 16;

  // Prefetch this wave's c_prev tile (16 rows x 256B = 32 lines; 1 line/lane)
  // so the epilogue loads hit L0/L2 after the WMMA chain.
  __builtin_prefetch(c_prev + (size_t)(rowBase + (lane >> 1)) * 64 + (lane & 1) * 32, 0, 1);

  // ---- stage f16 weights into LDS (coalesced uint4 copy of 64 KB) ----
  {
    const uint4* src = (const uint4*)Wt;
    uint4*       dst = (uint4*)ldsW;
#pragma unroll
    for (int i = 0; i < 16; ++i) {
      int idx = i * 256 + tid;         // 4096 uint4 total
      dst[idx] = src[idx];
    }
  }
  __syncthreads();

  // ---- accumulators initialized with gate biases ----
  v8f acc[16];
#pragma unroll
  for (int t = 0; t < 16; ++t) {
    const float* bp = (t < 4) ? b_i : (t < 8) ? b_f : (t < 12) ? b_o : b_g;
    float bv = bp[(t & 3) * 16 + colSel];
#pragma unroll
    for (int v = 0; v < 8; ++v) acc[t][v] = bv;
  }

  // ---- per-lane A-row setup ----
  const int arow = rowBase + colSel;         // A-matrix row for this lane
  const float mk = mask[arow];               // locked dropout scale

  // ---- GEMM: 4 K-chunks of 32, 16 N-tiles, 64 WMMAs ----
#pragma unroll
  for (int kb = 0; kb < 4; ++kb) {
    // Gather this lane's 16 A values straight from global (x for kb<2, h for kb>=2).
    const float* src = (kb < 2) ? (x_t + arow * 64 + kb * 32)
                                : (h_prev + arow * 64 + (kb - 2) * 32);
    const float* p = src + hi * 16;          // lanes 16-31 take the K+16 half
    float4 q0 = *(const float4*)(p);         // K +0..3   (or +16..19)
    float4 q1 = *(const float4*)(p + 4);     // K +4..7   (or +20..23)
    float4 q2 = *(const float4*)(p + 8);     // K +8..11  (or +24..27)
    float4 q3 = *(const float4*)(p + 12);    // K +12..15 (or +28..31)
    v16h afrag;
    afrag[0]  = (_Float16)(q0.x * mk); afrag[1]  = (_Float16)(q0.y * mk);
    afrag[2]  = (_Float16)(q0.z * mk); afrag[3]  = (_Float16)(q0.w * mk);
    afrag[4]  = (_Float16)(q1.x * mk); afrag[5]  = (_Float16)(q1.y * mk);
    afrag[6]  = (_Float16)(q1.z * mk); afrag[7]  = (_Float16)(q1.w * mk);
    afrag[8]  = (_Float16)(q2.x * mk); afrag[9]  = (_Float16)(q2.y * mk);
    afrag[10] = (_Float16)(q2.z * mk); afrag[11] = (_Float16)(q2.w * mk);
    afrag[12] = (_Float16)(q3.x * mk); afrag[13] = (_Float16)(q3.y * mk);
    afrag[14] = (_Float16)(q3.z * mk); afrag[15] = (_Float16)(q3.w * mk);

#pragma unroll
    for (int t = 0; t < 16; ++t) {
      // B fragment: column N = 16t + colSel, K = 32*kb + 16*hi + 0..15,
      // 16 contiguous f16 in ldsW -> 32B-aligned v16h (2x ds_load_b128).
      int n = t * 16 + colSel;
      const v16h bfrag = *(const v16h*)&ldsW[n * DH + kb * 32 + hi * 16];
      acc[t] = __builtin_amdgcn_wmma_f32_16x16x32_f16(
          false, afrag, false, bfrag, (short)0, acc[t], false, false);
    }
  }

  // ---- elementwise LSTM epilogue ----
  // D layout: lane covers rows M = v + 8*hi (v = VGPR index), col N = colSel (+16t).
#pragma unroll
  for (int t = 0; t < 4; ++t) {
    int j = t * 16 + colSel;                 // hidden index 0..63
#pragma unroll
    for (int v = 0; v < 8; ++v) {
      int row = rowBase + hi * 8 + v;
      float gi = acc[t][v];
      float gf = acc[t + 4][v];
      float go = acc[t + 8][v];
      float gg = acc[t + 12][v];
      float iv = sigmoid_f(gi);
      float fv = sigmoid_f(gf);
      float ov = sigmoid_f(go);
      float gv = tanh_f(gg);
      float cp = c_prev[(size_t)row * 64 + j];
      float cn = __builtin_fmaf(fv, cp, iv * gv);
      float hn = ov * tanh_f(cn);
      out[(size_t)row * 64 + j] = hn;                              // h
      out[(size_t)B_ROWS * 64 + (size_t)row * 64 + j] = cn;        // c
    }
  }
}

extern "C" void kernel_launch(void* const* d_in, const int* in_sizes, int n_in,
                              void* d_out, int out_size, void* d_ws, size_t ws_size,
                              hipStream_t stream) {
  const float* x_t    = (const float*)d_in[0];
  const float* h_prev = (const float*)d_in[1];
  const float* c_prev = (const float*)d_in[2];
  const float* mask   = (const float*)d_in[3];
  const float* W_i    = (const float*)d_in[4];
  const float* b_i    = (const float*)d_in[5];
  const float* W_f    = (const float*)d_in[6];
  const float* b_f    = (const float*)d_in[7];
  const float* W_o    = (const float*)d_in[8];
  const float* b_o    = (const float*)d_in[9];
  const float* W_g    = (const float*)d_in[10];
  const float* b_g    = (const float*)d_in[11];

  _Float16* Wt = (_Float16*)d_ws;          // 256*128 f16 = 64 KB scratch

  prep_weights<<<(NG * DH + 255) / 256, 256, 0, stream>>>(W_i, W_f, W_o, W_g, Wt);

  const size_t ldsBytes = (size_t)NG * DH * sizeof(_Float16);   // 64 KB
  lstm_cell<<<B_ROWS / 128, 256, ldsBytes, stream>>>(
      x_t, h_prev, c_prev, mask, b_i, b_f, b_o, b_g, Wt, (float*)d_out);
}
